// poMACEExp2MultiagentNetwork_87325275062417
// MI455X (gfx1250) — matrix operand order repaired and checked by reading, not compiled, not run on hardware
//
#include <hip/hip_runtime.h>
#include <hip/hip_bf16.h>

// ---------------------------------------------------------------------------
// poMACE multi-agent net: encoder -> GRU(T=512) -> linear+softmax
// A=8 B=128 T=512 IN=128 ENC=64 H=64 LAM=64 ACT=32
//  K1: fused enc=relu(X@Wenc+benc); gi=enc@Wih^T+bih (parallel, WMMA f16,
//      coalesced b128 stores of gi via LDS restage)
//  K2: sequential GRU: gh=h@Whh^T via WMMA (W_hh frags hoisted to registers,
//      h master copy in registers), gates f32 VALU
//  K3: logits=[h|noise]@Wout+bout via WMMA, softmax, b128 stores
// ---------------------------------------------------------------------------

typedef __attribute__((ext_vector_type(16))) _Float16 v16h;
typedef __attribute__((ext_vector_type(8)))  _Float16 v8h;
typedef __attribute__((ext_vector_type(8)))  float    v8f;

#define A_DIM   8
#define B_DIM   128
#define T_DIM   512
#define IN_DIM  128
#define ENC_DIM 64
#define H_DIM   64
#define LAM_DIM 64
#define ACT_DIM 32
#define NB      (A_DIM * B_DIM)          // 1024 batch rows for GRU
#define RROWS   ((long)NB * T_DIM)       // 524288 total rows

// ---- WMMA helpers (layouts per CDNA5 ISA 7.12.2) ----
__device__ __forceinline__ v8f wmma_f16(v16h a, v16h b, v8f c) {
  return __builtin_amdgcn_wmma_f32_16x16x32_f16(false, a, false, b, (short)0, c,
                                                false, false);
}

// A fragment: 16x32 f16 tile from row-major [16][K] image (LDS), k-slice k0.
// lane lo=M; halves 0..7 = k0+hi*8 .. +7 ; halves 8..15 = k0+16+hi*8 .. +7
__device__ __forceinline__ v16h load_afrag(const _Float16* src, int K, int k0) {
  const int lane = threadIdx.x & 31;
  const int lo = lane & 15, hi = lane >> 4;
  const _Float16* p = src + lo * K + k0 + hi * 8;
  union { v16h v; v8h h[2]; } u;
  u.h[0] = *(const v8h*)(p);
  u.h[1] = *(const v8h*)(p + 16);
  return u.v;
}

// B fragment: 32x16 (KxN) f16 tile from N-major [N][K] image (LDS).
// lane lo=N; halves 0..7 = k0+hi*8 .. +7 ; halves 8..15 = k0+16+hi*8 .. +7
__device__ __forceinline__ v16h load_bfrag(const _Float16* srcNT, int K, int n0,
                                           int k0) {
  const int lane = threadIdx.x & 31;
  const int lo = lane & 15, hi = lane >> 4;
  const _Float16* p = srcNT + (n0 + lo) * K + k0 + hi * 8;
  union { v16h v; v8h h[2]; } u;
  u.h[0] = *(const v8h*)(p);
  u.h[1] = *(const v8h*)(p + 16);
  return u.v;
}

__device__ __forceinline__ float fast_sigmoid(float x) {
  return 1.0f / (1.0f + __expf(-x));
}
__device__ __forceinline__ float fast_tanh(float x) {
  float e = __expf(-2.0f * x);
  return (1.0f - e) / (1.0f + e);
}

// ---------------------------------------------------------------------------
// Kernel 1: enc = relu(X @ Wenc + benc); gi = enc @ Wih^T + bih  -> f16 ws
// grid 4096 x 256 : 8 waves/block, one 16-row M-tile per wave (32768 tiles)
// sScratch[wave] (6KB) is reused: first as f16 X-tile (4KB), later as the
// gi tile staged for coalesced b128 global stores.
// ---------------------------------------------------------------------------
__global__ __launch_bounds__(256) void k_enc_gi(
    const float* __restrict__ x,     // [R][128]
    const float* __restrict__ Wenc,  // [128][64]  (K-major)
    const float* __restrict__ benc,  // [64]
    const float* __restrict__ Wih,   // [192][64]  (already N-major vs enc)
    const float* __restrict__ bih,   // [192]
    _Float16* __restrict__ gi)       // [R][192]
{
  __shared__ alignas(16) _Float16 sWencT[ENC_DIM * IN_DIM];   // [n][k] 16KB
  __shared__ alignas(16) _Float16 sWih[192 * ENC_DIM];        // [n][k] 24KB
  __shared__ float sBih[192];
  __shared__ float sBenc[ENC_DIM];
  __shared__ alignas(16) _Float16 sScratch[8][16 * 192];      // 48KB (reused)
  __shared__ alignas(16) _Float16 sE[8][16 * ENC_DIM];        // 16KB

  const int tid = threadIdx.x;
  const int wave = tid >> 5;
  const int lane = tid & 31;
  const int lo = lane & 15, hi = lane >> 4;

  // stage weights (Wenc transposed to N-major, Wih is already [N][K])
  for (int i = tid; i < IN_DIM * ENC_DIM; i += 256) {
    int k = i >> 6, n = i & 63;
    sWencT[n * IN_DIM + k] = (_Float16)Wenc[i];
  }
  for (int i = tid; i < 192 * ENC_DIM; i += 256) sWih[i] = (_Float16)Wih[i];
  if (tid < 192) sBih[tid] = bih[tid];
  if (tid < ENC_DIM) sBenc[tid] = benc[tid];

  const long tile = (long)blockIdx.x * 8 + wave;
  const long rBase = tile * 16;

  // stage this wave's 16x128 f32 rows -> f16 LDS (front of scratch)
  _Float16* aT = sScratch[wave];
  for (int i = lane; i < 512; i += 32) {  // 512 float4 chunks
    int row = i >> 5, c4 = (i & 31) * 4;
    float4 v = *(const float4*)(x + (rBase + row) * IN_DIM + c4);
    _Float16* d = aT + row * IN_DIM + c4;
    d[0] = (_Float16)v.x; d[1] = (_Float16)v.y;
    d[2] = (_Float16)v.z; d[3] = (_Float16)v.w;
  }
  __syncthreads();

  // ---- enc GEMM: [16x128] x [128x64], 4 k-slices, 4 n-tiles ----
  v16h aF0 = load_afrag(aT, IN_DIM, 0);
  v16h aF1 = load_afrag(aT, IN_DIM, 32);
  v16h aF2 = load_afrag(aT, IN_DIM, 64);
  v16h aF3 = load_afrag(aT, IN_DIM, 96);
  _Float16* eT = sE[wave];
#pragma unroll
  for (int n = 0; n < 4; ++n) {
    v8f acc = {};
    acc = wmma_f16(aF0, load_bfrag(sWencT, IN_DIM, n * 16, 0),  acc);
    acc = wmma_f16(aF1, load_bfrag(sWencT, IN_DIM, n * 16, 32), acc);
    acc = wmma_f16(aF2, load_bfrag(sWencT, IN_DIM, n * 16, 64), acc);
    acc = wmma_f16(aF3, load_bfrag(sWencT, IN_DIM, n * 16, 96), acc);
    const float bb = sBenc[n * 16 + lo];
#pragma unroll
    for (int p = 0; p < 8; ++p) {
      float v = acc[p] + bb;
      v = v > 0.0f ? v : 0.0f;  // relu
      eT[(p + hi * 8) * ENC_DIM + n * 16 + lo] = (_Float16)v;
    }
  }
  __syncthreads();  // eT visible; also fences off scratch reuse below

  // ---- gi GEMM: [16x64] x [64x192], 2 k-slices, 12 n-tiles ----
  v16h gF0 = load_afrag(eT, ENC_DIM, 0);
  v16h gF1 = load_afrag(eT, ENC_DIM, 32);
  _Float16* giT = sScratch[wave];  // 16x192 f16 tile, row-major
#pragma unroll
  for (int n = 0; n < 12; ++n) {
    v8f acc = {};
    acc = wmma_f16(gF0, load_bfrag(sWih, ENC_DIM, n * 16, 0),  acc);
    acc = wmma_f16(gF1, load_bfrag(sWih, ENC_DIM, n * 16, 32), acc);
    const float bb = sBih[n * 16 + lo];
#pragma unroll
    for (int p = 0; p < 8; ++p)
      giT[(p + hi * 8) * 192 + n * 16 + lo] = (_Float16)(acc[p] + bb);
  }
  __syncthreads();

  // coalesced store: 16 rows x 192 halves = one contiguous 6KB global span
  _Float16* gdst = gi + rBase * 192;
  for (int i = lane; i < 384; i += 32)        // 384 v8h chunks
    *(v8h*)(gdst + i * 8) = *(const v8h*)(giT + i * 8);
}

// ---------------------------------------------------------------------------
// Kernel 2: GRU recurrence. grid 64 x 128. 16 batch rows / WG.
// W_hh B-fragments hoisted to registers; h master copy in registers;
// per step: 6 wmma/wave + gate math; gi loads issued before the WMMAs.
// ---------------------------------------------------------------------------
__global__ __launch_bounds__(128) void k_gru(
    const float* __restrict__ h0,     // [NB][64]
    const float* __restrict__ Whh,    // [192][64]  N-major
    const float* __restrict__ bhh,    // [192]
    const _Float16* __restrict__ gi,  // [NB*T][192] (includes bih)
    _Float16* __restrict__ hseq)      // [NB*T][64]
{
  __shared__ alignas(16) _Float16 sWhh[192 * H_DIM];  // 24KB
  __shared__ float sBhh[192];
  __shared__ alignas(16) _Float16 sH16[16 * H_DIM];   // h as f16 for WMMA
  __shared__ alignas(16) float sGh[16 * 192];         // gh scratch 12KB

  const int tid = threadIdx.x;
  const int wave = tid >> 5, lane = tid & 31;
  const int lo = lane & 15, hi = lane >> 4;
  const int n0 = blockIdx.x * 16;

  for (int i = tid; i < 192 * H_DIM; i += 128) sWhh[i] = (_Float16)Whh[i];
  if (tid < 128) sBhh[tid] = bhh[tid];
  if (tid < 64)  sBhh[128 + tid] = bhh[128 + tid];

  const int row = tid >> 3;            // 0..15
  const int c0 = (tid & 7) * 8;        // 0,8,..,56

  // h master copy lives in registers (each thread owns row, c0..c0+7)
  float hreg[8];
  {
    v8h h16;
#pragma unroll
    for (int j = 0; j < 8; ++j) {
      hreg[j] = h0[(n0 + row) * H_DIM + c0 + j];
      h16[j] = (_Float16)hreg[j];
    }
    *(v8h*)(sH16 + row * H_DIM + c0) = h16;
  }
  __syncthreads();

  // hoist loop-invariant W_hh fragments (3 n-tiles x 2 k-slices per wave)
  v16h bF[3][2];
#pragma unroll
  for (int j = 0; j < 3; ++j) {
    bF[j][0] = load_bfrag(sWhh, H_DIM, (wave * 3 + j) * 16, 0);
    bF[j][1] = load_bfrag(sWhh, H_DIM, (wave * 3 + j) * 16, 32);
  }
  // hoist per-thread b_hh values
  float bhr[8], bhz[8], bhn[8];
#pragma unroll
  for (int j = 0; j < 8; ++j) {
    bhr[j] = sBhh[c0 + j];
    bhz[j] = sBhh[64 + c0 + j];
    bhn[j] = sBhh[128 + c0 + j];
  }

  const long rgBase = (long)(n0 + row) * T_DIM;
  for (int t = 0; t < T_DIM; ++t) {
    // issue gi loads early so their latency hides behind the WMMAs
    const _Float16* gp = gi + (rgBase + t) * 192;
    v8h g_r = *(const v8h*)(gp + c0);
    v8h g_z = *(const v8h*)(gp + 64 + c0);
    v8h g_n = *(const v8h*)(gp + 128 + c0);

    // gh = h @ Whh^T : each wave does 3 n-tiles
    v16h aF0 = load_afrag(sH16, H_DIM, 0);
    v16h aF1 = load_afrag(sH16, H_DIM, 32);
#pragma unroll
    for (int j = 0; j < 3; ++j) {
      const int nt = wave * 3 + j;
      v8f acc = {};
      acc = wmma_f16(aF0, bF[j][0], acc);
      acc = wmma_f16(aF1, bF[j][1], acc);
#pragma unroll
      for (int p = 0; p < 8; ++p)
        sGh[(p + hi * 8) * 192 + nt * 16 + lo] = acc[p];
    }
    __syncthreads();

    // gates: 8 contiguous hidden units per thread, vectorized gh reads
    const float* ghp = sGh + row * 192 + c0;
    float4 vr0 = *(const float4*)(ghp);
    float4 vr1 = *(const float4*)(ghp + 4);
    float4 vz0 = *(const float4*)(ghp + 64);
    float4 vz1 = *(const float4*)(ghp + 68);
    float4 vn0 = *(const float4*)(ghp + 128);
    float4 vn1 = *(const float4*)(ghp + 132);
    float ghr[8] = {vr0.x, vr0.y, vr0.z, vr0.w, vr1.x, vr1.y, vr1.z, vr1.w};
    float ghz[8] = {vz0.x, vz0.y, vz0.z, vz0.w, vz1.x, vz1.y, vz1.z, vz1.w};
    float ghn[8] = {vn0.x, vn0.y, vn0.z, vn0.w, vn1.x, vn1.y, vn1.z, vn1.w};

    v8h hv;
#pragma unroll
    for (int j = 0; j < 8; ++j) {
      float r = fast_sigmoid((float)g_r[j] + ghr[j] + bhr[j]);
      float z = fast_sigmoid((float)g_z[j] + ghz[j] + bhz[j]);
      float nn = fast_tanh((float)g_n[j] + r * (ghn[j] + bhn[j]));
      float hNew = (1.0f - z) * nn + z * hreg[j];
      hreg[j] = hNew;
      hv[j] = (_Float16)hNew;
    }
    *(v8h*)(sH16 + row * H_DIM + c0) = hv;              // one ds_store_b128
    *(v8h*)(hseq + (rgBase + t) * H_DIM + c0) = hv;     // one global b128
    __syncthreads();
  }
}

// ---------------------------------------------------------------------------
// Kernel 3: logits = [hseq | noise] @ Wout + bout ; softmax(32) -> f32 out
// grid 4096 x 256 : one 16-row tile per wave.
// ---------------------------------------------------------------------------
__global__ __launch_bounds__(256) void k_out(
    const _Float16* __restrict__ hseq,  // [R][64]
    const float* __restrict__ noise,    // [R][64]
    const float* __restrict__ Wout,     // [128][32] K-major
    const float* __restrict__ bout,     // [32]
    float* __restrict__ probs)          // [R][32]
{
  __shared__ alignas(16) _Float16 sWoutT[ACT_DIM * 128];  // [n][k] 8KB
  __shared__ float sBout[ACT_DIM];
  __shared__ alignas(16) _Float16 sA[8][16 * 128];        // 32KB
  __shared__ alignas(16) float sLog[128 * ACT_DIM];       // 16KB

  const int tid = threadIdx.x, wave = tid >> 5, lane = tid & 31;
  const int lo = lane & 15, hi = lane >> 4;

  for (int i = tid; i < 128 * ACT_DIM; i += 256) {
    int k = i >> 5, n = i & 31;
    sWoutT[n * 128 + k] = (_Float16)Wout[i];
  }
  if (tid < ACT_DIM) sBout[tid] = bout[tid];

  const long tile = (long)blockIdx.x * 8 + wave;
  const long rBase = tile * 16;
  _Float16* aT = sA[wave];
  // h part: 16 rows x 64 halves = 128 v8h chunks
  for (int i = lane; i < 128; i += 32) {
    int row = i >> 3, c8 = (i & 7) * 8;
    *(v8h*)(aT + row * 128 + c8) =
        *(const v8h*)(hseq + (rBase + row) * H_DIM + c8);
  }
  // noise part: 16 rows x 64 floats = 256 float4 chunks -> f16
  for (int i = lane; i < 256; i += 32) {
    int row = i >> 4, c4 = (i & 15) * 4;
    float4 v = *(const float4*)(noise + (rBase + row) * LAM_DIM + c4);
    _Float16* d = aT + row * 128 + 64 + c4;
    d[0] = (_Float16)v.x; d[1] = (_Float16)v.y;
    d[2] = (_Float16)v.z; d[3] = (_Float16)v.w;
  }
  __syncthreads();

  v16h aF0 = load_afrag(aT, 128, 0);
  v16h aF1 = load_afrag(aT, 128, 32);
  v16h aF2 = load_afrag(aT, 128, 64);
  v16h aF3 = load_afrag(aT, 128, 96);
#pragma unroll
  for (int n = 0; n < 2; ++n) {
    v8f acc = {};
    acc = wmma_f16(aF0, load_bfrag(sWoutT, 128, n * 16, 0),  acc);
    acc = wmma_f16(aF1, load_bfrag(sWoutT, 128, n * 16, 32), acc);
    acc = wmma_f16(aF2, load_bfrag(sWoutT, 128, n * 16, 64), acc);
    acc = wmma_f16(aF3, load_bfrag(sWoutT, 128, n * 16, 96), acc);
    const float bb = sBout[n * 16 + lo];
#pragma unroll
    for (int p = 0; p < 8; ++p)
      sLog[(wave * 16 + p + hi * 8) * ACT_DIM + n * 16 + lo] = acc[p] + bb;
  }
  __syncthreads();

  if (tid < 128) {
    const float* lrow = sLog + tid * ACT_DIM;
    float v[ACT_DIM];
#pragma unroll
    for (int i = 0; i < 8; ++i) {
      float4 q = *(const float4*)(lrow + i * 4);
      v[i * 4 + 0] = q.x; v[i * 4 + 1] = q.y;
      v[i * 4 + 2] = q.z; v[i * 4 + 3] = q.w;
    }
    float m = -1e30f;
#pragma unroll
    for (int i = 0; i < ACT_DIM; ++i) m = fmaxf(m, v[i]);
    float s = 0.0f;
#pragma unroll
    for (int i = 0; i < ACT_DIM; ++i) { v[i] = __expf(v[i] - m); s += v[i]; }
    const float inv = 1.0f / s;
    const long rg = (long)blockIdx.x * 128 + tid;
    float* op = probs + rg * ACT_DIM;
#pragma unroll
    for (int i = 0; i < 8; ++i) {
      float4 o;
      o.x = v[i * 4 + 0] * inv; o.y = v[i * 4 + 1] * inv;
      o.z = v[i * 4 + 2] * inv; o.w = v[i * 4 + 3] * inv;
      *(float4*)(op + i * 4) = o;
    }
  }
}

// ---------------------------------------------------------------------------
extern "C" void kernel_launch(void* const* d_in, const int* in_sizes, int n_in,
                              void* d_out, int out_size, void* d_ws,
                              size_t ws_size, hipStream_t stream) {
  const float* agent_inputs = (const float*)d_in[0];  // [A,B,T,IN]
  const float* hidden_init  = (const float*)d_in[1];  // [A,B,H]
  const float* noise_x      = (const float*)d_in[2];  // [A,B,T,LAM]
  const float* W_enc        = (const float*)d_in[3];  // [IN,ENC]
  const float* b_enc        = (const float*)d_in[4];  // [ENC]
  const float* W_ih         = (const float*)d_in[5];  // [3H,ENC]
  const float* b_ih         = (const float*)d_in[6];  // [3H]
  const float* W_hh         = (const float*)d_in[7];  // [3H,H]
  const float* b_hh         = (const float*)d_in[8];  // [3H]
  const float* W_out        = (const float*)d_in[9];  // [H+LAM,ACT]
  const float* b_out        = (const float*)d_in[10]; // [ACT]
  float* probs = (float*)d_out;

  // workspace: gi [R][192] f16, then hseq [R][64] f16
  _Float16* gi   = (_Float16*)d_ws;
  _Float16* hseq = (_Float16*)((char*)d_ws + RROWS * 192 * sizeof(_Float16));

  // K1: 32768 M-tiles of 16 rows, 8 tiles per 256-thread block
  k_enc_gi<<<4096, 256, 0, stream>>>(agent_inputs, W_enc, b_enc, W_ih, b_ih, gi);
  // K2: 1024 batch rows / 16 per WG
  k_gru<<<64, 128, 0, stream>>>(hidden_init, W_hh, b_hh, gi, hseq);
  // K3: output + softmax
  k_out<<<4096, 256, 0, stream>>>(hseq, noise_x, W_out, b_out, probs);
}
